// DualTowerHypergraphSummariser_49864570307270
// MI455X (gfx1250) — matrix operand order repaired and checked by reading, not compiled, not run on hardware
//
#include <hip/hip_runtime.h>
#include <hip/hip_bf16.h>
#include <math.h>

// Problem constants (match reference)
#define BB 64
#define SS 128
#define NN 100
#define DD 1024
#define NDm 768
#define BN (BB*NN)

typedef __attribute__((ext_vector_type(16))) _Float16 v16h;
typedef __attribute__((ext_vector_type(8)))  float    v8f;
typedef unsigned int v4u __attribute__((ext_vector_type(4)));
typedef int v8i32 __attribute__((ext_vector_type(8)));
typedef int v4i32 __attribute__((ext_vector_type(4)));

// GEMM tiling
#define TM 64
#define TN 128
#define TK 32
// LDS row stride in halves: 40 halves = 80B. Conflict-free for 16-row fragment
// reads (m*20 mod 64 distinct), 16B aligned, and reproducible by TDM padding
// (pad_interval=3 -> every 16 DWORDs, pad_amount=3 -> +4 DWORDs).
#define LD_S 40

#if defined(__has_builtin)
#if __has_builtin(__builtin_amdgcn_tensor_load_to_lds) && __has_builtin(__builtin_amdgcn_s_wait_tensorcnt)
#define USE_TDM 1
#endif
#endif
#ifndef USE_TDM
#define USE_TDM 0
#endif

__device__ __forceinline__ float sigmf(float x) { return 1.f / (1.f + expf(-x)); }
__device__ __forceinline__ float geluf(float x) { return 0.5f * x * (1.f + erff(x * 0.70710678118654752f)); }

// ---- WMMA fragment loaders (layouts per CDNA5 ISA 7.12.2, wave32) ----
__device__ __forceinline__ v16h frag_a(const _Float16* sm, int lane) {
  union { v16h h; unsigned int u[8]; } f;
  const unsigned int* row = (const unsigned int*)(sm + (lane & 15) * LD_S);
  int kh = (lane >> 4) << 3;
#pragma unroll
  for (int v = 0; v < 8; ++v) {
    int k0 = ((v >> 2) << 4) + kh + ((v & 3) << 1);
    f.u[v] = row[k0 >> 1];
  }
  return f.h;
}
__device__ __forceinline__ v16h frag_b(const _Float16* sm, int lane) {
  union { v16h h; unsigned int u[8]; } f;
  const unsigned int* row = (const unsigned int*)(sm + (lane & 15) * LD_S);
  int kh = (lane >> 4) << 3;
#pragma unroll
  for (int v = 0; v < 8; ++v) f.u[v] = row[(((v >> 2) << 4) + kh + ((v & 3) << 1)) >> 1];
  return f.h;
}

#if USE_TDM
// Issue a TDM 2D tile load: tileRows x tileK halves, row stride strideElems,
// zero-fill past (remK, remRows). LDS gets padded rows of LD_S halves.
__device__ __forceinline__ void tdm_issue(unsigned lds_off, const _Float16* g,
                                          int tileK, int tileRows, long strideElems,
                                          long remK, long remRows) {
  unsigned long long ga = (unsigned long long)(size_t)g;
  if (remK < 0) remK = 0;
  if (remRows < 0) remRows = 0;
  v4u g0;
  g0.x = 1u;                                    // count=1 (valid descriptor)
  g0.y = lds_off;                               // lds_addr (bytes)
  g0.z = (unsigned)ga;                          // global_addr[31:0]
  g0.w = (unsigned)((ga >> 32) & 0x1FFFFFFu) | (2u << 30);  // addr[56:32] | type=2
  unsigned long long t0 = (unsigned long long)remK;
  unsigned long long t1 = (unsigned long long)remRows;
  v8i32 g1;
  g1[0] = (int)((1u << 16)      // data_size = 1 -> 2 bytes
              | (1u << 20)      // pad_enable
              | (3u << 22)      // pad_interval: 16 DWORDs
              | (3u << 25));    // pad_amount: 4 DWORDs
  g1[1] = (int)((unsigned)(t0 & 0xFFFFu) << 16);                       // dim0[15:0]
  g1[2] = (int)(((t0 >> 16) & 0xFFFFu) | ((t1 & 0xFFFFu) << 16));      // dim0[31:16]|dim1[15:0]
  g1[3] = (int)(((t1 >> 16) & 0xFFFFu) | (((unsigned)tileK) << 16));   // dim1[31:16]|tile_dim0
  g1[4] = (int)((unsigned)tileRows & 0xFFFFu);                         // tile_dim1 (tile_dim2=0)
  g1[5] = (int)((unsigned long long)strideElems & 0xFFFFFFFFu);        // dim0_stride[31:0]
  g1[6] = (int)(((unsigned long long)strideElems >> 32) & 0xFFFFu);    // dim0_stride[47:32]
  g1[7] = 0;
  v4i32 z4 = {0, 0, 0, 0};
  v8i32 z8 = {0, 0, 0, 0, 0, 0, 0, 0};
  __builtin_amdgcn_tensor_load_to_lds(g0, g1, z4, z4, z8, 0);
}
#endif

// ---------------- Weight GEMM: C[M,O] = op( A16[M,K] * W16[O,K]^T ) ----------------
// result = (acc + bias)*outScale ; optional exact-gelu ; optional += Cin
__global__ __launch_bounds__(256)
void wgemm_kernel(const _Float16* __restrict__ A, long lda,
                  const _Float16* __restrict__ W,
                  const float* __restrict__ bias,
                  const float* __restrict__ Cin,
                  float* __restrict__ C, long ldc,
                  int M, int K, int O, int gelu_flag, float outScale) {
  __shared__ alignas(16) _Float16 smA[2][TM * LD_S];
  __shared__ alignas(16) _Float16 smB[2][TN * LD_S];
  const int tid = threadIdx.x;
  const int lane = tid & 31;
  const int wave = tid >> 5;
  const int wm = wave & 3;
  const int wn = wave >> 2;
  const long row0 = (long)blockIdx.x * TM;
  const long col0 = (long)blockIdx.y * TN;
  const _Float16* Abase = A + row0 * lda;
  const _Float16* Wbase = W + col0 * (long)K;
  const int nk = (K + TK - 1) / TK;

  v8f acc[4] = {};

#if USE_TDM
  const bool issuer = (tid < 32);
  unsigned offA[2] = { (unsigned)(size_t)&smA[0][0], (unsigned)(size_t)&smA[1][0] };
  unsigned offB[2] = { (unsigned)(size_t)&smB[0][0], (unsigned)(size_t)&smB[1][0] };
  if (issuer) {
    tdm_issue(offA[0], Abase, TK, TM, lda, K, M - row0);
    tdm_issue(offB[0], Wbase, TK, TN, K, K, O - col0);
  }
  for (int i = 0; i < nk; ++i) {
    if (issuer) __builtin_amdgcn_s_wait_tensorcnt(0);
    __syncthreads();
    if (issuer && i + 1 < nk) {
      int kb = (i + 1) * TK;
      tdm_issue(offA[(i + 1) & 1], Abase + kb, TK, TM, lda, K - kb, M - row0);
      tdm_issue(offB[(i + 1) & 1], Wbase + kb, TK, TN, K, K - kb, O - col0);
    }
    const _Float16* sa = &smA[i & 1][0];
    const _Float16* sb = &smB[i & 1][0];
    v16h a = frag_a(sa + wm * 16 * LD_S, lane);
#pragma unroll
    for (int nt = 0; nt < 4; ++nt) {
      v16h b = frag_b(sb + (wn * 64 + nt * 16) * LD_S, lane);
      acc[nt] = __builtin_amdgcn_wmma_f32_16x16x32_f16(
          false, a, false, b, (short)0, acc[nt], false, false);
    }
    __syncthreads();
  }
#else
  for (int i = 0; i < nk; ++i) {
    int kb = i * TK;
    for (int idx = tid; idx < TM * TK; idx += 256) {
      int r = idx >> 5, c = idx & 31;
      long gr = row0 + r; int gk = kb + c;
      smA[0][r * LD_S + c] = (gr < M && gk < K) ? Abase[(long)r * lda + gk] : (_Float16)0;
    }
    for (int idx = tid; idx < TN * TK; idx += 256) {
      int n = idx >> 5, c = idx & 31;
      long gn = col0 + n; int gk = kb + c;
      smB[0][n * LD_S + c] = (gn < O && gk < K) ? Wbase[(long)n * K + gk] : (_Float16)0;
    }
    __syncthreads();
    v16h a = frag_a(&smA[0][0] + wm * 16 * LD_S, lane);
#pragma unroll
    for (int nt = 0; nt < 4; ++nt) {
      v16h b = frag_b(&smB[0][0] + (wn * 64 + nt * 16) * LD_S, lane);
      acc[nt] = __builtin_amdgcn_wmma_f32_16x16x32_f16(
          false, a, false, b, (short)0, acc[nt], false, false);
    }
    __syncthreads();
  }
#endif

  const int mloc = (lane >> 4) << 3;
  const int nloc = lane & 15;
#pragma unroll
  for (int nt = 0; nt < 4; ++nt) {
    long col = col0 + wn * 64 + nt * 16 + nloc;
#pragma unroll
    for (int r = 0; r < 8; ++r) {
      long row = row0 + wm * 16 + r + mloc;
      if (row < M) {
        float v = acc[nt][r];
        if (bias) v += bias[col];
        if (gelu_flag) v = geluf(v);
        v *= outScale;
        if (Cin) v += Cin[row * ldc + col];
        C[row * ldc + col] = v;
      }
    }
  }
}

// ---- Batched GEMM: C16[b][M,Nc] = A16[b][M,K] * B16[b][K,Nc] (row-major B) ----
// A tile via TDM (k-contiguous); B tile needs a transpose -> cooperative load.
__global__ __launch_bounds__(256)
void bgemm_kernel(const _Float16* __restrict__ A, long strideA, long lda,
                  const _Float16* __restrict__ Bm, long strideB, long ldb,
                  _Float16* __restrict__ C, long strideC, long ldc,
                  int M, int K, int Nc) {
  __shared__ alignas(16) _Float16 smA[TM * LD_S];
  __shared__ alignas(16) _Float16 smB[TN * LD_S];
  const int tid = threadIdx.x;
  const int lane = tid & 31;
  const int wave = tid >> 5;
  const int wm = wave & 3;
  const int wn = wave >> 2;
  const long row0 = (long)blockIdx.x * TM;
  const long col0 = (long)blockIdx.y * TN;
  const int bz = blockIdx.z;
  const _Float16* Ab = A + (long)bz * strideA + row0 * lda;
  const _Float16* Bb = Bm + (long)bz * strideB;
  _Float16* Cb = C + (long)bz * strideC;
  const int nk = (K + TK - 1) / TK;
#if USE_TDM
  const bool issuer = (tid < 32);
  unsigned offA0 = (unsigned)(size_t)&smA[0];
#endif

  v8f acc[4] = {};
  for (int i = 0; i < nk; ++i) {
    int kb = i * TK;
#if USE_TDM
    if (issuer) tdm_issue(offA0, Ab + kb, TK, TM, lda, K - kb, M - row0);
#else
    for (int idx = tid; idx < TM * TK; idx += 256) {
      int r = idx >> 5, c = idx & 31;
      long gr = row0 + r; int gk = kb + c;
      smA[r * LD_S + c] = (gr < M && gk < K) ? Ab[(long)r * lda + gk] : (_Float16)0;
    }
#endif
    // B tile: transpose [k][n] -> LDS [n][k]; consecutive threads read consecutive n
    for (int idx = tid; idx < TN * TK; idx += 256) {
      int n = idx & (TN - 1);
      int k = idx >> 7;
      long gk = kb + k;
      _Float16 v = (_Float16)0;
      if (gk < K) v = Bb[gk * ldb + col0 + n];
      smB[n * LD_S + k] = v;
    }
#if USE_TDM
    if (issuer) __builtin_amdgcn_s_wait_tensorcnt(0);
#endif
    __syncthreads();
    v16h a = frag_a(&smA[0] + wm * 16 * LD_S, lane);
#pragma unroll
    for (int nt = 0; nt < 4; ++nt) {
      v16h b = frag_b(&smB[0] + (wn * 64 + nt * 16) * LD_S, lane);
      acc[nt] = __builtin_amdgcn_wmma_f32_16x16x32_f16(
          false, a, false, b, (short)0, acc[nt], false, false);
    }
    __syncthreads();
  }
  const int mloc = (lane >> 4) << 3;
  const int nloc = lane & 15;
#pragma unroll
  for (int nt = 0; nt < 4; ++nt) {
    long col = col0 + wn * 64 + nt * 16 + nloc;
    if (col >= Nc) continue;
#pragma unroll
    for (int r = 0; r < 8; ++r) {
      long row = row0 + wm * 16 + r + mloc;
      if (row < M) Cb[row * ldc + col] = (_Float16)acc[nt][r];
    }
  }
}

// ---------------- small fused kernels ----------------
__global__ void zero_kernel(float* p, long n) {
  long i = (long)blockIdx.x * blockDim.x + threadIdx.x;
  if (i < n) p[i] = 0.f;
}
__global__ void cvt_kernel(const float* __restrict__ in, _Float16* __restrict__ out, long n) {
  long i = (long)blockIdx.x * blockDim.x + threadIdx.x;
  if (i < n) out[i] = (_Float16)in[i];
}

// row LayerNorm over D=1024: optional residual; f32 out + up to two f16 outs
__global__ __launch_bounds__(256)
void ln_kernel(const float* __restrict__ X, long ldx,
               const float* __restrict__ R, long ldr,
               const float* __restrict__ g, const float* __restrict__ bta,
               float* __restrict__ Yf, long ldyf,
               _Float16* __restrict__ Yh, long ldyh,
               _Float16* __restrict__ Yh2, long ldyh2) {
  __shared__ float red[256];
  const int tid = threadIdx.x;
  const long row = blockIdx.x;
  const float* x = X + row * ldx;
  const float* r = R ? (R + row * ldr) : nullptr;
  float vals[4], s = 0.f;
#pragma unroll
  for (int i = 0; i < 4; ++i) {
    int d = tid + i * 256;
    float v = x[d] + (r ? r[d] : 0.f);
    vals[i] = v; s += v;
  }
  red[tid] = s; __syncthreads();
  for (int st = 128; st > 0; st >>= 1) { if (tid < st) red[tid] += red[tid + st]; __syncthreads(); }
  float mean = red[0] * (1.f / DD);
  __syncthreads();
  float vs = 0.f;
#pragma unroll
  for (int i = 0; i < 4; ++i) { float dlt = vals[i] - mean; vs += dlt * dlt; }
  red[tid] = vs; __syncthreads();
  for (int st = 128; st > 0; st >>= 1) { if (tid < st) red[tid] += red[tid + st]; __syncthreads(); }
  float inv = rsqrtf(red[0] * (1.f / DD) + 1e-5f);
#pragma unroll
  for (int i = 0; i < 4; ++i) {
    int d = tid + i * 256;
    float o = (vals[i] - mean) * inv * g[d] + bta[d];
    if (Yf)  Yf[row * ldyf + d] = o;
    if (Yh)  Yh[row * ldyh + d] = (_Float16)o;
    if (Yh2) Yh2[row * ldyh2 + d] = (_Float16)o;
  }
}

// h0 = (type_emb[id] + LN(nameProj)) * mask ; writes f32 h and f16 copy
__global__ __launch_bounds__(256)
void init_h_kernel(const float* __restrict__ nameProj,
                   const float* __restrict__ ng, const float* __restrict__ nb,
                   const float* __restrict__ type_emb, const int* __restrict__ ids,
                   const float* __restrict__ mask, float* __restrict__ h,
                   _Float16* __restrict__ h16) {
  __shared__ float red[256];
  const int tid = threadIdx.x;
  const long bn = blockIdx.x;
  const float* x = nameProj + bn * DD;
  float vals[4], s = 0.f;
#pragma unroll
  for (int i = 0; i < 4; ++i) { float v = x[tid + i * 256]; vals[i] = v; s += v; }
  red[tid] = s; __syncthreads();
  for (int st = 128; st > 0; st >>= 1) { if (tid < st) red[tid] += red[tid + st]; __syncthreads(); }
  float mean = red[0] * (1.f / DD);
  __syncthreads();
  float vs = 0.f;
#pragma unroll
  for (int i = 0; i < 4; ++i) { float dlt = vals[i] - mean; vs += dlt * dlt; }
  red[tid] = vs; __syncthreads();
  for (int st = 128; st > 0; st >>= 1) { if (tid < st) red[tid] += red[tid + st]; __syncthreads(); }
  float inv = rsqrtf(red[0] * (1.f / DD) + 1e-5f);
  int t = ids[bn];
  float m = mask[bn];
#pragma unroll
  for (int i = 0; i < 4; ++i) {
    int d = tid + i * 256;
    float ln = (vals[i] - mean) * inv * ng[d] + nb[d];
    float o = (type_emb[(long)t * DD + d] + ln) * m;
    h[bn * DD + d] = o;
    h16[bn * DD + d] = (_Float16)o;
  }
}

// roles[b,n] = inc[b,n,s]*mask ; mean16[b,d] = sum_n roles*h / max(sum roles, 1)
__global__ __launch_bounds__(256)
void roles_mean_kernel(const float* __restrict__ inc, const float* __restrict__ mask,
                       const float* __restrict__ h, float* __restrict__ rolesG,
                       _Float16* __restrict__ mean16, int s) {
  __shared__ float roles_s[NN];
  __shared__ float red[256];
  __shared__ float ws_s;
  const int b = blockIdx.x, tid = threadIdx.x;
  float part = 0.f;
  for (int n = tid; n < NN; n += 256) {
    float r = inc[((long)b * NN + n) * SS + s] * mask[(long)b * NN + n];
    roles_s[n] = r; rolesG[(long)b * NN + n] = r; part += r;
  }
  red[tid] = part; __syncthreads();
  for (int st = 128; st > 0; st >>= 1) { if (tid < st) red[tid] += red[tid + st]; __syncthreads(); }
  if (tid == 0) ws_s = fmaxf(red[0], 1.f);
  __syncthreads();
  float ws = ws_s;
  for (int d = tid; d < DD; d += 256) {
    float acc = 0.f;
    for (int n = 0; n < NN; ++n) acc += roles_s[n] * h[((long)b * NN + n) * DD + d];
    mean16[(long)b * DD + d] = (_Float16)(acc / ws);
  }
}

// logits = geluH @ sg_W2^T + b2 ; softmax -> gate[b,3]
__global__ void gate_kernel(const float* __restrict__ H, const float* __restrict__ W2,
                            const float* __restrict__ b2, float* __restrict__ gate) {
  __shared__ float red[3][32];
  const int b = blockIdx.x, tid = threadIdx.x;
  for (int o = 0; o < 3; ++o) {
    float p = 0.f;
    for (int k = tid; k < 256; k += 32) p += H[(long)b * 256 + k] * W2[o * 256 + k];
    red[o][tid] = p;
  }
  __syncthreads();
  if (tid == 0) {
    float l[3];
    for (int o = 0; o < 3; ++o) {
      float s = 0.f;
      for (int i = 0; i < 32; ++i) s += red[o][i];
      l[o] = s + b2[o];
    }
    float mx = fmaxf(l[0], fmaxf(l[1], l[2]));
    float e0 = expf(l[0] - mx), e1 = expf(l[1] - mx), e2 = expf(l[2] - mx);
    float sum = e0 + e1 + e2;
    gate[b * 3 + 0] = e0 / sum; gate[b * 3 + 1] = e1 / sum; gate[b * 3 + 2] = e2 / sum;
  }
}

// arc16[b,n,j] = inc*role*decay(j<s), normalized over j
__global__ void arc_kernel(const float* __restrict__ inc, const float* __restrict__ roles,
                           _Float16* __restrict__ arc16, int s) {
  __shared__ float red[SS];
  const long bn = blockIdx.x; const int j = threadIdx.x;
  float role = roles[bn];
  float v = 0.f;
  if (j < s) {
    float denom = fmaxf((float)s - 1.f, 1.f);
    v = inc[bn * SS + j] * role * expf(-1.f + (float)j / denom);
  }
  red[j] = v; __syncthreads();
  for (int st = 64; st > 0; st >>= 1) { if (j < st) red[j] += red[j + st]; __syncthreads(); }
  arc16[bn * SS + j] = (_Float16)(v / fmaxf(red[0], 1e-8f));
}

// co16[b,n,m] = roles[n]*roles[m]*(m!=n), normalized over m
__global__ void co_kernel(const float* __restrict__ roles, _Float16* __restrict__ co16) {
  __shared__ float red[128];
  const long idx = blockIdx.x;                 // b*NN + n
  const long b = idx / NN; const int n = (int)(idx % NN);
  const int m = threadIdx.x;
  float v = 0.f;
  if (m < NN && m != n) v = roles[b * NN + n] * roles[b * NN + m];
  red[m] = v; __syncthreads();
  for (int st = 64; st > 0; st >>= 1) { if (m < st) red[m] += red[m + st]; __syncthreads(); }
  if (m < NN) co16[idx * NN + m] = (_Float16)(v / fmaxf(red[0], 1e-8f));
}

// msg16 = g0*msg_scene + g1*msgA + g2*msgI + role*role_w*in_scene  (f16 out)
__global__ __launch_bounds__(256)
void msg_fuse_kernel(const float* __restrict__ msgS, const float* __restrict__ msgA,
                     const float* __restrict__ msgI, const float* __restrict__ gate,
                     const float* __restrict__ roles, const float* __restrict__ role_w,
                     _Float16* __restrict__ msg16) {
  const long bn = blockIdx.x; const long b = bn / NN;
  const int tid = threadIdx.x;
  float g0 = gate[b * 3], g1 = gate[b * 3 + 1], g2 = gate[b * 3 + 2];
  float role = roles[bn];
  float ins = role > 0.f ? 1.f : 0.f;
#pragma unroll
  for (int i = 0; i < 4; ++i) {
    int d = tid + i * 256;
    float v = g0 * msgS[b * DD + d] + g1 * msgA[bn * DD + d]
            + g2 * msgI[bn * DD + d] + role * role_w[d] * ins;
    msg16[bn * DD + d] = (_Float16)v;
  }
}

// GRU combine + LN + clip + conditional update of h (f32 + f16 copies)
__global__ __launch_bounds__(256)
void gru_fuse_kernel(const float* __restrict__ gi, const float* __restrict__ gh,
                     const float* __restrict__ roles, const float* __restrict__ ug,
                     const float* __restrict__ ub, float* __restrict__ h,
                     _Float16* __restrict__ h16) {
  __shared__ float red[256];
  const long bn = blockIdx.x; const int tid = threadIdx.x;
  const float role = roles[bn];
  const bool ins = role > 0.f;
  const float* gib = gi + bn * 3L * DD;
  const float* ghb = gh + bn * 3L * DD;
  float hnew[4], hprev[4], s = 0.f;
#pragma unroll
  for (int i = 0; i < 4; ++i) {
    int d = tid + i * 256;
    float ir = gib[d], iz = gib[DD + d], inn = gib[2 * DD + d];
    float hr = ghb[d], hz = ghb[DD + d], hn = ghb[2 * DD + d];
    float r = sigmf(ir + hr), z = sigmf(iz + hz);
    float nn = tanhf(inn + r * hn);
    float hp = h[bn * DD + d]; hprev[i] = hp;
    float v = (1.f - z) * nn + z * hp;
    hnew[i] = v; s += v;
  }
  red[tid] = s; __syncthreads();
  for (int st = 128; st > 0; st >>= 1) { if (tid < st) red[tid] += red[tid + st]; __syncthreads(); }
  float mean = red[0] * (1.f / DD);
  __syncthreads();
  float vs = 0.f;
#pragma unroll
  for (int i = 0; i < 4; ++i) { float dlt = hnew[i] - mean; vs += dlt * dlt; }
  red[tid] = vs; __syncthreads();
  for (int st = 128; st > 0; st >>= 1) { if (tid < st) red[tid] += red[tid + st]; __syncthreads(); }
  float inv = rsqrtf(red[0] * (1.f / DD) + 1e-5f);
#pragma unroll
  for (int i = 0; i < 4; ++i) {
    int d = tid + i * 256;
    float o = (hnew[i] - mean) * inv * ug[d] + ub[d];
    o = fminf(fmaxf(o, -50.f), 50.f);
    float out = ins ? o : hprev[i];
    h[bn * DD + d] = out;
    h16[bn * DD + d] = (_Float16)out;
  }
}

// ---------------- host orchestration ----------------
static inline dim3 wgrid(int M, int O) { return dim3((unsigned)((M + TM - 1) / TM), (unsigned)(O / TN)); }
static inline unsigned cdiv(long n, int b) { return (unsigned)((n + b - 1) / b); }

extern "C" void kernel_launch(void* const* d_in, const int* in_sizes, int n_in,
                              void* d_out, int out_size, void* d_ws, size_t ws_size,
                              hipStream_t stream) {
  (void)in_sizes; (void)n_in; (void)out_size; (void)ws_size;
  const float* scene  = (const float*)d_in[0];
  const float* inc    = (const float*)d_in[1];
  const float* mask   = (const float*)d_in[2];
  const float* nameE  = (const float*)d_in[3];
  const float* type_emb = (const float*)d_in[4];
  const float* name_W = (const float*)d_in[5];
  const float* name_g = (const float*)d_in[6];
  const float* name_b = (const float*)d_in[7];
  const float* ne_W = (const float*)d_in[8];
  const float* ne_b = (const float*)d_in[9];
  const float* te_W = (const float*)d_in[10];
  const float* te_b = (const float*)d_in[11];
  const float* en_g = (const float*)d_in[12];
  const float* en_b = (const float*)d_in[13];
  const float* sg_W1 = (const float*)d_in[14];
  const float* sg_b1 = (const float*)d_in[15];
  const float* sg_W2 = (const float*)d_in[16];
  const float* sg_b2 = (const float*)d_in[17];
  const float* ms_W = (const float*)d_in[18];
  const float* ms_b = (const float*)d_in[19];
  const float* ma_W = (const float*)d_in[20];
  const float* ma_b = (const float*)d_in[21];
  const float* mi_W = (const float*)d_in[22];
  const float* mi_b = (const float*)d_in[23];
  const float* role_w = (const float*)d_in[24];
  const float* gru_Wih = (const float*)d_in[25];
  const float* gru_Whh = (const float*)d_in[26];
  const float* gru_bih = (const float*)d_in[27];
  const float* gru_bhh = (const float*)d_in[28];
  const float* upd_g = (const float*)d_in[29];
  const float* upd_b = (const float*)d_in[30];
  const float* eo_W = (const float*)d_in[31];
  const float* eo_b = (const float*)d_in[32];
  const float* eo_g = (const float*)d_in[33];
  const float* eo_b2 = (const float*)d_in[34];
  const float* res_g = (const float*)d_in[35];
  const float* res_b = (const float*)d_in[36];
  const int* type_ids = (const int*)d_in[37];

  char* base = (char*)d_ws;
  size_t off = 0;
  auto allocF = [&](size_t n) { float* p = (float*)(base + off); off = (off + n * 4 + 255) & ~(size_t)255; return p; };
  auto allocH = [&](size_t n) { _Float16* p = (_Float16*)(base + off); off = (off + n * 2 + 255) & ~(size_t)255; return p; };

  // f32 state
  float* E     = allocF((size_t)BB * SS * DD);
  float* tmpE  = allocF((size_t)BB * DD);
  float* msgS  = allocF((size_t)BB * DD);
  float* gateH = allocF((size_t)BB * 256);
  float* gate  = allocF((size_t)BB * 4);
  float* roles = allocF((size_t)BB * NN);
  float* msgA  = allocF((size_t)BN * DD);
  float* msgI  = allocF((size_t)BN * DD);
  float* gi    = allocF((size_t)BN * 3 * DD);
  float* gh    = allocF((size_t)BN * 3 * DD);
  float* nameProj = allocF((size_t)BN * DD);
  float* Hg = gi;                           // final-pass scratch reuses gi
  // f16 weights (converted once)
  _Float16* nameWh = allocH((size_t)DD * NDm);
  _Float16* neWh = allocH((size_t)DD * DD);
  _Float16* teWh = allocH((size_t)DD * DD);
  _Float16* sgW1h = allocH((size_t)256 * DD);
  _Float16* msWh = allocH((size_t)DD * DD);
  _Float16* maWh = allocH((size_t)DD * DD);
  _Float16* miWh = allocH((size_t)DD * DD);
  _Float16* Wihh = allocH((size_t)3 * DD * DD);
  _Float16* Whhh = allocH((size_t)3 * DD * DD);
  _Float16* eoWh = allocH((size_t)DD * DD);
  // f16 activations
  _Float16* scene16 = allocH((size_t)BB * SS * DD);
  _Float16* name16  = allocH((size_t)BN * NDm);
  _Float16* buf16   = allocH((size_t)BB * SS * DD);
  _Float16* h16     = allocH((size_t)BN * DD);
  _Float16* e16     = allocH((size_t)BB * DD);
  _Float16* mean16  = allocH((size_t)BB * DD);
  _Float16* arc16   = allocH((size_t)BB * NN * SS);
  _Float16* co16    = allocH((size_t)BB * NN * NN);
  _Float16* aggA16  = allocH((size_t)BN * DD);
  _Float16* aggI16  = allocH((size_t)BN * DD);
  _Float16* msg16   = allocH((size_t)BN * DD);
  _Float16* Hln16   = allocH((size_t)BB * SS * DD);

  float* H = (float*)d_out;
  float* h = H + (size_t)BB * SS * DD;       // h_final lives in d_out

  const long SD = (long)SS * DD;

  // one-time conversions to f16 (weights + static activations) and buf16 = 0
  {
    struct { const float* s; _Float16* d; long n; } cv[] = {
      { name_W, nameWh, (long)DD * NDm }, { ne_W, neWh, (long)DD * DD },
      { te_W, teWh, (long)DD * DD },      { sg_W1, sgW1h, (long)256 * DD },
      { ms_W, msWh, (long)DD * DD },      { ma_W, maWh, (long)DD * DD },
      { mi_W, miWh, (long)DD * DD },      { gru_Wih, Wihh, (long)3 * DD * DD },
      { gru_Whh, Whhh, (long)3 * DD * DD }, { eo_W, eoWh, (long)DD * DD },
      { scene, scene16, (long)BB * SS * DD }, { nameE, name16, (long)BN * NDm },
    };
    for (auto& c : cv) cvt_kernel<<<cdiv(c.n, 256), 256, 0, stream>>>(c.s, c.d, c.n);
    long nb = (long)BB * SS * DD / 2;  // zero buf16 as floats
    zero_kernel<<<cdiv(nb, 256), 256, 0, stream>>>((float*)buf16, nb);
  }

  // h0
  wgemm_kernel<<<wgrid(BN, DD), 256, 0, stream>>>(name16, NDm, nameWh, nullptr, nullptr,
                                                  nameProj, DD, BN, NDm, DD, 0, 1.f);
  init_h_kernel<<<BN, 256, 0, stream>>>(nameProj, name_g, name_b, type_emb, type_ids, mask, h, h16);

  for (int s = 0; s < SS; ++s) {
    roles_mean_kernel<<<BB, 256, 0, stream>>>(inc, mask, h, roles, mean16, s);
    // e = LN(mean@ne_W^T + ne_b + scene_s@te_W^T + te_b)
    wgemm_kernel<<<wgrid(BB, DD), 256, 0, stream>>>(mean16, DD, neWh, ne_b, nullptr,
                                                    tmpE, DD, BB, DD, DD, 0, 1.f);
    wgemm_kernel<<<wgrid(BB, DD), 256, 0, stream>>>(scene16 + (size_t)s * DD, SD, teWh, te_b, tmpE,
                                                    tmpE, DD, BB, DD, DD, 0, 1.f);
    ln_kernel<<<BB, 256, 0, stream>>>(tmpE, DD, nullptr, 0, en_g, en_b,
                                      E + (size_t)s * DD, SD, e16, DD, buf16 + (size_t)s * DD, SD);
    // gate
    wgemm_kernel<<<wgrid(BB, 256), 256, 0, stream>>>(e16, DD, sgW1h, sg_b1, nullptr,
                                                     gateH, 256, BB, DD, 256, 1, 1.f);
    gate_kernel<<<BB, 32, 0, stream>>>(gateH, sg_W2, sg_b2, gate);
    // msg_scene
    wgemm_kernel<<<wgrid(BB, DD), 256, 0, stream>>>(e16, DD, msWh, ms_b, nullptr,
                                                    msgS, DD, BB, DD, DD, 0, 1.f);
    // arc stream
    arc_kernel<<<BB * NN, SS, 0, stream>>>(inc, roles, arc16, s);
    bgemm_kernel<<<dim3(2, DD / TN, BB), 256, 0, stream>>>(arc16, (long)NN * SS, SS,
                                                           buf16, SD, DD,
                                                           aggA16, (long)NN * DD, DD, NN, SS, DD);
    wgemm_kernel<<<wgrid(BN, DD), 256, 0, stream>>>(aggA16, DD, maWh, ma_b, nullptr,
                                                    msgA, DD, BN, DD, DD, 0, s > 0 ? 1.f : 0.f);
    // interaction stream
    co_kernel<<<BB * NN, 128, 0, stream>>>(roles, co16);
    bgemm_kernel<<<dim3(2, DD / TN, BB), 256, 0, stream>>>(co16, (long)NN * NN, NN,
                                                           h16, (long)NN * DD, DD,
                                                           aggI16, (long)NN * DD, DD, NN, NN, DD);
    wgemm_kernel<<<wgrid(BN, DD), 256, 0, stream>>>(aggI16, DD, miWh, mi_b, nullptr,
                                                    msgI, DD, BN, DD, DD, 0, 1.f);
    // combine + GRU
    msg_fuse_kernel<<<BN, 256, 0, stream>>>(msgS, msgA, msgI, gate, roles, role_w, msg16);
    wgemm_kernel<<<wgrid(BN, 3 * DD), 256, 0, stream>>>(msg16, DD, Wihh, gru_bih, nullptr,
                                                        gi, 3 * DD, BN, DD, 3 * DD, 0, 1.f);
    wgemm_kernel<<<wgrid(BN, 3 * DD), 256, 0, stream>>>(h16, DD, Whhh, gru_bhh, nullptr,
                                                        gh, 3 * DD, BN, DD, 3 * DD, 0, 1.f);
    gru_fuse_kernel<<<BN, 256, 0, stream>>>(gi, gh, roles, upd_g, upd_b, h, h16);
  }

  // H = LN(gelu(LN(E + scene) @ eo_W^T + eo_b))
  ln_kernel<<<BB * SS, 256, 0, stream>>>(E, DD, scene, DD, res_g, res_b,
                                         nullptr, 0, Hln16, DD, nullptr, 0);
  wgemm_kernel<<<wgrid(BB * SS, DD), 256, 0, stream>>>(Hln16, DD, eoWh, eo_b, nullptr,
                                                       Hg, DD, BB * SS, DD, DD, 1, 1.f);
  ln_kernel<<<BB * SS, 256, 0, stream>>>(Hg, DD, nullptr, 0, eo_g, eo_b2,
                                         H, DD, nullptr, 0, nullptr, 0);
}
